// DUGC_39213051412672
// MI455X (gfx1250) — matrix operand order repaired and checked by reading, not compiled, not run on hardware
//
#include <hip/hip_runtime.h>
#include <math.h>

typedef __attribute__((ext_vector_type(16))) _Float16 v16h;
typedef __attribute__((ext_vector_type(8)))  float    v8f;

// Problem constants (from reference setup_inputs)
constexpr int BB = 8;
constexpr int CC = 384;
constexpr int HH = 64;
constexpr int WW = 64;
constexpr int NN = HH * WW;        // 4096
constexpr int KNBR = 8;
constexpr int EE = NN * KNBR;      // 32768
constexpr int MR = BB * NN;        // 32768 rows for GEMMs

#define DEVINL __device__ __forceinline__

// K-swizzle for the f16 A-operand shadow: within each 32-wide K block,
// reorder 8-element chunks 0,1,2,3 -> positions 0,2,1,3 so that each WMMA
// lane's A fragment (ISA 7.12.2: K = {h*8..h*8+7} ++ {16+h*8..16+h*8+7})
// becomes one contiguous 32-byte run at offset k0 + h*16.
DEVINL int kswz(int k) {
    int pos = k & 31;
    int chunk = pos >> 3;                                  // 0..3
    int nchunk = ((chunk & 1) << 1) | (chunk >> 1);        // 0,2,1,3
    return (k & ~31) | (nchunk << 3) | (pos & 7);
}

// ---------------------------------------------------------------------------
// fea_mean[n, c] = mean over b of fea[b, c, n]
__global__ void k_mean(const float* __restrict__ fea, float* __restrict__ fm) {
    int idx = blockIdx.x * blockDim.x + threadIdx.x;
    if (idx >= NN * CC) return;
    int n = idx % NN;
    int c = idx / NN;
    float s = 0.f;
    #pragma unroll
    for (int b = 0; b < BB; ++b)
        s += fea[((size_t)b * CC + c) * NN + n];
    fm[(size_t)n * CC + c] = s * (1.0f / BB);
}

// rn2[n] = ||fm[n,:]||^2
__global__ void k_rn2(const float* __restrict__ fm, float* __restrict__ rn2) {
    int n = blockIdx.x * blockDim.x + threadIdx.x;
    if (n >= NN) return;
    const float* row = fm + (size_t)n * CC;
    float s = 0.f;
    for (int c = 0; c < CC; ++c) s += row[c] * row[c];
    rn2[n] = s;
}

// ---------------------------------------------------------------------------
// stable top-8 insertion: keep ascending by (d, idx); ties -> smaller idx
DEVINL void topk_insert(float* bd, int* bi, float d, int j) {
    if (d > bd[7] || (d == bd[7] && j >= bi[7])) return;
    bd[7] = d; bi[7] = j;
    #pragma unroll
    for (int t = 7; t > 0; --t) {
        bool sw = (bd[t] < bd[t-1]) || (bd[t] == bd[t-1] && bi[t] < bi[t-1]);
        if (sw) {
            float td = bd[t]; bd[t] = bd[t-1]; bd[t-1] = td;
            int   ti = bi[t]; bi[t] = bi[t-1]; bi[t-1] = ti;
        }
    }
}

// one block (256 thr) per row i: combined distance + top-8 (smallest)
__global__ void k_topk(const float* __restrict__ fm, const float* __restrict__ rn2,
                       int* __restrict__ tk) {
    __shared__ float s_fi[CC];
    __shared__ float s_d[256 * 8];
    __shared__ int   s_i[256 * 8];
    int i = blockIdx.x;
    int tid = threadIdx.x;
    for (int c = tid; c < CC; c += 256) s_fi[c] = fm[(size_t)i * CC + c];
    __syncthreads();

    float iy = (float)(i / WW), ix = (float)(i % WW);
    float r2i = rn2[i];
    float bd[8]; int bi[8];
    #pragma unroll
    for (int t = 0; t < 8; ++t) { bd[t] = 3.4e38f; bi[t] = 0x7FFFFFFF; }

    const float4* sf4 = (const float4*)s_fi;
    for (int j = tid; j < NN; j += 256) {
        const float4* fj4 = (const float4*)(fm + (size_t)j * CC);
        float dot = 0.f;
        #pragma unroll 4
        for (int c4 = 0; c4 < CC / 4; ++c4) {
            float4 a = sf4[c4], b = fj4[c4];
            dot += a.x * b.x + a.y * b.y + a.z * b.z + a.w * b.w;
        }
        float fd2 = r2i + rn2[j] - 2.f * dot;
        float fd = sqrtf(fmaxf(fd2, 0.f));
        float dy = iy - (float)(j / WW);
        float dx = ix - (float)(j % WW);
        float sd = sqrtf(dy * dy + dx * dx);
        float d = 0.7f * sd + 0.3f * fd;
        topk_insert(bd, bi, d, j);
    }
    #pragma unroll
    for (int t = 0; t < 8; ++t) { s_d[tid * 8 + t] = bd[t]; s_i[tid * 8 + t] = bi[t]; }
    __syncthreads();
    if (tid == 0) {
        float md[8]; int mi[8];
        #pragma unroll
        for (int t = 0; t < 8; ++t) { md[t] = 3.4e38f; mi[t] = 0x7FFFFFFF; }
        for (int q = 0; q < 256 * 8; ++q) topk_insert(md, mi, s_d[q], s_i[q]);
        #pragma unroll
        for (int t = 0; t < 8; ++t) tk[(size_t)i * 8 + t] = mi[t];
    }
}

// ---------------------------------------------------------------------------
// in-degree (deterministic: each node scans all edges)
__global__ void k_count(const int* __restrict__ tk, int* __restrict__ deg) {
    int j = blockIdx.x * blockDim.x + threadIdx.x;
    if (j >= NN) return;
    int cnt = 0;
    for (int e = 0; e < EE; ++e) cnt += (tk[e] == j);
    deg[j] = cnt;
}

__global__ void k_scan(const int* __restrict__ deg, int* __restrict__ roff) {
    if (blockIdx.x == 0 && threadIdx.x == 0) {
        int acc = 0;
        for (int j = 0; j < NN; ++j) { roff[j] = acc; acc += deg[j]; }
        roff[NN] = acc;
    }
}

// reverse lists (sources of incoming edges), deterministic edge order
__global__ void k_fill(const int* __restrict__ tk, const int* __restrict__ roff,
                       int* __restrict__ rsrc) {
    int j = blockIdx.x * blockDim.x + threadIdx.x;
    if (j >= NN) return;
    int p = roff[j];
    for (int e = 0; e < EE; ++e)
        if (tk[e] == j) rsrc[p++] = e >> 3;   // src = e / KNBR
}

__global__ void k_dinv(const int* __restrict__ deg, float* __restrict__ dinv) {
    int j = blockIdx.x * blockDim.x + threadIdx.x;
    if (j >= NN) return;
    dinv[j] = rsqrtf((float)deg[j]);
}

// x[b, n, c] = fea[b, c, n]  (f32 + K-swizzled f16 shadow for WMMA)
__global__ void k_xinit(const float* __restrict__ fea, float* __restrict__ x,
                        _Float16* __restrict__ x16) {
    int idx = blockIdx.x * blockDim.x + threadIdx.x;
    if (idx >= BB * NN * CC) return;
    int c = idx % CC;
    int n = (idx / CC) % NN;
    int b = idx / (CC * NN);
    float v = fea[((size_t)b * CC + c) * NN + n];
    x[idx] = v;
    x16[(size_t)(idx - c) + kswz(c)] = (_Float16)v;
}

// transpose + convert weight: Wt[n, kswz(k)] = (f16) W[k, n]   (W is [K, Nn] row-major)
__global__ void k_wt(const float* __restrict__ Wsrc, _Float16* __restrict__ Wt,
                     int K, int Nn) {
    int idx = blockIdx.x * blockDim.x + threadIdx.x;
    if (idx >= K * Nn) return;
    int k = idx % K;
    int n = idx / K;
    Wt[(size_t)n * K + kswz(k)] = (_Float16)Wsrc[(size_t)k * Nn + n];
}

// ---------------------------------------------------------------------------
// WMMA f16 GEMM:  D[M,Nn] = A16s[M,K] @ Bt16^T
//   A16s : [M,K] f16, K-swizzled (kswz) -> lane frag is contig 32B at k0+h*16
//   Bt16 : [Nn,K] f16 (weights transposed) -> lane frag is contig 32B at k0+h*16
// Each wave computes a 16 x (16*JT) strip; K stepped by 32; no spills, no guards.
template <int JT>
__global__ void k_gemm_f16(const _Float16* __restrict__ A16,
                           const _Float16* __restrict__ Bt16,
                           float* __restrict__ D, int M, int Nn, int K) {
    int wave = threadIdx.x >> 5;
    int lane = threadIdx.x & 31;
    int ntiles16 = Nn >> 4;
    int nstrips = ntiles16 / JT;                 // exact for all shapes used
    int totWaves = (M >> 4) * nstrips;
    int id = blockIdx.x * (blockDim.x >> 5) + wave;
    if (id >= totWaves) return;                  // wave-uniform exit
    int mt = id / nstrips;
    int ns = id - mt * nstrips;
    int m0 = mt << 4;
    int h = lane >> 4;
    int t = lane & 15;

    v8f acc[JT] = {};
    const _Float16* ap = A16 + (size_t)(m0 + t) * K + (h << 4);
    const _Float16* bp = Bt16 + ((size_t)((ns * JT) << 4) + t) * K + (h << 4);

    for (int k0 = 0; k0 < K; k0 += 32) {
        v16h av = *(const v16h*)(ap + k0);
        #pragma unroll
        for (int j = 0; j < JT; ++j) {
            v16h bv = *(const v16h*)(bp + (size_t)(j << 4) * K + k0);
            acc[j] = __builtin_amdgcn_wmma_f32_16x16x32_f16(
                false, av, false, bv, (short)0, acc[j], false, false);
        }
    }
    // C/D 16x16 f32 layout: VGPR v -> row m0 + v + 8*h, col n0 + t
    #pragma unroll
    for (int j = 0; j < JT; ++j) {
        int ncol = ((ns * JT + j) << 4) + t;
        #pragma unroll
        for (int v = 0; v < 8; ++v)
            D[(size_t)(m0 + v + (h << 3)) * Nn + ncol] = acc[j][v];
    }
}

// ---------------------------------------------------------------------------
// xo[b,j,c] = x[b,j,c] + relu( bias[c] + sum_{e in rev[j]} dinv[src]*dinv[j]*xw[b,src,c] )
// also emits K-swizzled f16 shadow for the next WMMA GEMM
__global__ void k_agg(const float* __restrict__ x, const float* __restrict__ xw,
                      const float* __restrict__ bias, const int* __restrict__ rsrc,
                      const int* __restrict__ roff, const float* __restrict__ dinv,
                      float* __restrict__ xo, _Float16* __restrict__ xo16) {
    int idx = blockIdx.x * blockDim.x + threadIdx.x;
    if (idx >= BB * NN * CC) return;
    int c = idx % CC;
    int j = (idx / CC) % NN;
    int b = idx / (CC * NN);
    float dj = dinv[j];
    float s = 0.f;
    int e0 = roff[j], e1 = roff[j + 1];
    for (int e = e0; e < e1; ++e) {
        int i = rsrc[e];
        s += dinv[i] * dj * xw[((size_t)b * NN + i) * CC + c];
    }
    float v = s + bias[c];
    v = v > 0.f ? v : 0.f;
    float r = x[idx] + v;
    xo[idx] = r;
    xo16[(size_t)(idx - c) + kswz(c)] = (_Float16)r;
}

// ---------------------------------------------------------------------------
DEVINL float gelu_tanh(float x) {
    float x3 = x * x * x;
    float tt = tanhf(0.7978845608028654f * (x + 0.044715f * x3));
    return 0.5f * x * (1.f + tt);
}

__global__ void k_bias_gelu(float* __restrict__ h, const float* __restrict__ bias,
                            _Float16* __restrict__ h16, int total, int cols) {
    int idx = blockIdx.x * blockDim.x + threadIdx.x;
    if (idx >= total) return;
    int c = idx % cols;
    float v = gelu_tanh(h[idx] + bias[c]);
    h[idx] = v;
    h16[(size_t)(idx - c) + kswz(c)] = (_Float16)v;
}

// unc[m] = sigmoid( h2[m,:] . U3 + ub3 )   (96 -> 1)
__global__ void k_dot_sig(const float* __restrict__ h2, const float* __restrict__ U3,
                          const float* __restrict__ ub3, float* __restrict__ unc) {
    int m = blockIdx.x * blockDim.x + threadIdx.x;
    if (m >= MR) return;
    const float* row = h2 + (size_t)m * 96;
    float s = ub3[0];
    #pragma unroll 4
    for (int c = 0; c < 96; ++c) s += row[c] * U3[c];
    unc[m] = 1.f / (1.f + expf(-s));
}

// out[b,c,n] = fea[b,c,n] * (1 + unc[b,n])
__global__ void k_final(const float* __restrict__ fea, const float* __restrict__ unc,
                        float* __restrict__ out) {
    int idx = blockIdx.x * blockDim.x + threadIdx.x;
    if (idx >= BB * CC * NN) return;
    int n = idx % NN;
    int b = idx / (CC * NN);
    out[idx] = fea[idx] * (1.f + unc[(size_t)b * NN + n]);
}

// ---------------------------------------------------------------------------
extern "C" void kernel_launch(void* const* d_in, const int* in_sizes, int n_in,
                              void* d_out, int out_size, void* d_ws, size_t ws_size,
                              hipStream_t stream) {
    (void)in_sizes; (void)n_in; (void)out_size; (void)ws_size;
    const float* fea = (const float*)d_in[0];
    const float* W1  = (const float*)d_in[1];
    const float* b1  = (const float*)d_in[2];
    const float* W2  = (const float*)d_in[3];
    const float* b2  = (const float*)d_in[4];
    const float* W3  = (const float*)d_in[5];
    const float* b3  = (const float*)d_in[6];
    const float* U1  = (const float*)d_in[7];
    const float* ub1 = (const float*)d_in[8];
    const float* U2  = (const float*)d_in[9];
    const float* ub2 = (const float*)d_in[10];
    const float* U3  = (const float*)d_in[11];
    const float* ub3 = (const float*)d_in[12];
    float* out = (float*)d_out;

    // workspace carve-out
    size_t off = 0;
    auto alloc = [&](size_t bytes) -> void* {
        void* p = (void*)((char*)d_ws + off);
        off += (bytes + 255) & ~(size_t)255;
        return p;
    };
    float*     fm   = (float*)    alloc((size_t)NN * CC * 4);
    float*     rn2  = (float*)    alloc((size_t)NN * 4);
    int*       tk   = (int*)      alloc((size_t)EE * 4);
    int*       deg  = (int*)      alloc((size_t)NN * 4);
    float*     dinv = (float*)    alloc((size_t)NN * 4);
    int*       roff = (int*)      alloc((size_t)(NN + 1) * 4);
    int*       rsrc = (int*)      alloc((size_t)EE * 4);
    float*     xA   = (float*)    alloc((size_t)MR * CC * 4);
    float*     xB   = (float*)    alloc((size_t)MR * CC * 4);
    float*     xw   = (float*)    alloc((size_t)MR * CC * 4);   // GEMM f32 out / h1
    _Float16*  x16  = (_Float16*) alloc((size_t)MR * CC * 2);   // f16 shadow (K-swizzled)
    _Float16*  wt1  = (_Float16*) alloc((size_t)CC * CC * 2);
    _Float16*  wt2  = (_Float16*) alloc((size_t)CC * CC * 2);
    _Float16*  wt3  = (_Float16*) alloc((size_t)CC * CC * 2);
    _Float16*  ut1  = (_Float16*) alloc((size_t)192 * CC * 2);  // [192, 384]
    _Float16*  ut2  = (_Float16*) alloc((size_t)96 * 192 * 2);  // [96, 192]
    float*     unc  = (float*)    alloc((size_t)MR * 4);

    const int TPB = 256;
    auto cdiv = [](int a, int b) { return (a + b - 1) / b; };
    auto gemmBlocks = [&](int M, int Nn, int jt) {
        int nstrips = (Nn >> 4) / jt;
        return cdiv((M >> 4) * nstrips, TPB / 32);
    };

    // graph build
    k_mean <<<cdiv(NN * CC, TPB), TPB, 0, stream>>>(fea, fm);
    k_rn2  <<<cdiv(NN, TPB),      TPB, 0, stream>>>(fm, rn2);
    k_topk <<<NN,                 TPB, 0, stream>>>(fm, rn2, tk);
    k_count<<<cdiv(NN, TPB),      TPB, 0, stream>>>(tk, deg);
    k_scan <<<1, 1, 0, stream>>>(deg, roff);
    k_fill <<<cdiv(NN, TPB),      TPB, 0, stream>>>(tk, roff, rsrc);
    k_dinv <<<cdiv(NN, TPB),      TPB, 0, stream>>>(deg, dinv);
    k_xinit<<<cdiv(BB * NN * CC, TPB), TPB, 0, stream>>>(fea, xA, x16);

    // weight pre-transpose + f16 convert
    k_wt<<<cdiv(CC * CC, TPB),  TPB, 0, stream>>>(W1, wt1, CC, CC);
    k_wt<<<cdiv(CC * CC, TPB),  TPB, 0, stream>>>(W2, wt2, CC, CC);
    k_wt<<<cdiv(CC * CC, TPB),  TPB, 0, stream>>>(W3, wt3, CC, CC);
    k_wt<<<cdiv(CC * 192, TPB), TPB, 0, stream>>>(U1, ut1, CC, 192);
    k_wt<<<cdiv(192 * 96, TPB), TPB, 0, stream>>>(U2, ut2, 192, 96);

    // 3 GCN layers (WMMA GEMM + gather aggregate), ping-pong xA/xB
    const _Float16* Wts[3] = { wt1, wt2, wt3 };
    const float*    bs[3]  = { b1, b2, b3 };
    float* xc = xA;
    float* xn = xB;
    for (int l = 0; l < 3; ++l) {
        k_gemm_f16<4><<<gemmBlocks(MR, CC, 4), TPB, 0, stream>>>(x16, Wts[l], xw, MR, CC, CC);
        k_agg<<<cdiv(BB * NN * CC, TPB), TPB, 0, stream>>>(xc, xw, bs[l], rsrc, roff,
                                                           dinv, xn, x16);
        float* tmp = xc; xc = xn; xn = tmp;
    }
    // now xc = xB (final GCN f32), x16 = f16 shadow of it, xn = xA (free)

    // MLP head
    k_gemm_f16<4><<<gemmBlocks(MR, 192, 4), TPB, 0, stream>>>(x16, ut1, xw, MR, 192, CC);
    k_bias_gelu<<<cdiv(MR * 192, TPB), TPB, 0, stream>>>(xw, ub1, x16, MR * 192, 192);

    float* h2 = xn;
    k_gemm_f16<3><<<gemmBlocks(MR, 96, 3), TPB, 0, stream>>>(x16, ut2, h2, MR, 96, 192);
    k_bias_gelu<<<cdiv(MR * 96, TPB), TPB, 0, stream>>>(h2, ub2, x16, MR * 96, 96);

    k_dot_sig<<<cdiv(MR, TPB), TPB, 0, stream>>>(h2, U3, ub3, unc);
    k_final<<<cdiv(BB * CC * NN, TPB), TPB, 0, stream>>>(fea, unc, out);
}
